// HyperGTConv_63771674411308
// MI455X (gfx1250) — matrix operand order repaired and checked by reading, not compiled, not run on hardware
//
#include <hip/hip_runtime.h>
#include <hip/hip_bf16.h>

// ---------------------------------------------------------------------------
// Performer attention (HyperGTConv) for MI455X / gfx1250, wave32 + bf16 WMMA.
// proj is folded into Wq/Wk so the feature map is a single fused GEMM + exp.
// ---------------------------------------------------------------------------

typedef __attribute__((ext_vector_type(16))) __bf16 v16bf;
typedef __attribute__((ext_vector_type(8)))  float  v8f;

#define WMMA_BF16(A, B, C) \
  __builtin_amdgcn_wmma_f32_16x16x32_bf16(false, (A), false, (B), (short)0, (C), false, false)

constexpr int    kN    = 50000;
constexpr int    kC    = 256;          // input channels
constexpr int    kHD   = 512;          // H*D = H*M
constexpr int    kH    = 8;
constexpr int    kD    = 64;
constexpr int    kNCH  = (kN + 31) / 32;  // 1563 node-chunks of 32
constexpr int    kNTW  = kHD / 16;     // 32 column tiles in the big weights
constexpr float  kEPS  = 1e-6f;
// ratio * data_normalizer = (1/sqrt(64)) * 64^(-1/4)
constexpr float  kScale = 0.04419417382415922f;

union FragU { uint4 q[2]; v16bf v; };

__device__ inline v8f v8zero() {
  v8f z;
#pragma unroll
  for (int i = 0; i < 8; ++i) z[i] = 0.f;
  return z;
}

// Packed fragment: 32 bytes per lane, contiguous.  tile -> base + tile*1KB.
__device__ inline v16bf load_packed_frag(const __bf16* base, long tile, int lane) {
  FragU u;
  const uint4* p = (const uint4*)(base + ((size_t)tile * 32 + lane) * 16);
  u.q[0] = p[0];
  u.q[1] = p[1];
  return u.v;
}

// A-fragment (16x32 bf16) from a row-major matrix; `row` already points at
// this lane's row.  Lanes 0-15 hold K {0..7,16..23}, lanes 16-31 {8..15,24..31}.
__device__ inline v16bf load_afrag_row(const __bf16* row, int kbase, int hi) {
  FragU u;
  u.q[0] = *(const uint4*)(row + kbase + hi * 8);
  u.q[1] = *(const uint4*)(row + kbase + hi * 8 + 16);
  return u.v;
}

// ---------------------------------------------------------------------------
__global__ void k_zero(float* p, int n) {
  int i = blockIdx.x * blockDim.x + threadIdx.x;
  if (i < n) p[i] = 0.f;
}

// W~ = scale * W @ (I_H (x) proj^T), written in packed-B fragment layout.
// Also computes the fused bias b~.
__global__ void k_prep_wproj(const float* __restrict__ W, const float* __restrict__ b,
                             const float* __restrict__ proj,
                             __bf16* __restrict__ Wp, float* __restrict__ bt) {
  int id = blockIdx.x * blockDim.x + threadIdx.x;
  if (id < kC * kHD) {
    int c = id / kHD, j = id % kHD;
    int h = j / kD, m = j % kD;
    float acc = 0.f;
    for (int d = 0; d < kD; ++d)
      acc += W[c * kHD + h * kD + d] * proj[m * kD + d];
    acc *= kScale;
    int kt = c >> 5, klocal = c & 31, khalf = (klocal >> 3) & 1;
    int nt = j >> 4, nl = j & 15;
    int lane = khalf * 16 + nl, t = (klocal & 7) + ((klocal >> 4) << 3);
    Wp[((size_t)(kt * kNTW + nt) * 32 + lane) * 16 + t] = (__bf16)acc;
  }
  if (id < kHD) {
    int h = id / kD, m = id % kD;
    float acc = 0.f;
    for (int d = 0; d < kD; ++d)
      acc += b[h * kD + d] * proj[m * kD + d];
    bt[id] = acc * kScale;
  }
}

// Plain bf16 conversion + packed-B layout (for Wv [256x512] and Wo [512x64]).
__global__ void k_pack_w(const float* __restrict__ W, __bf16* __restrict__ Wp,
                         int K, int Ncols) {
  int id = blockIdx.x * blockDim.x + threadIdx.x;
  if (id >= K * Ncols) return;
  int k = id / Ncols, j = id % Ncols;
  int nt_total = Ncols >> 4;
  int kt = k >> 5, klocal = k & 31, khalf = (klocal >> 3) & 1;
  int nt = j >> 4, nl = j & 15;
  int lane = khalf * 16 + nl, t = (klocal & 7) + ((klocal >> 4) << 3);
  Wp[((size_t)(kt * nt_total + nt) * 32 + lane) * 16 + t] = (__bf16)W[id];
}

// ---------------------------------------------------------------------------
// Pass A: per 32-node chunk compute Qp=exp(x@Wq~+bq~)+eps (row-major bf16),
// Kp (repacked to WMMA A-fragment layout, K = node dim) and V (repacked to
// WMMA B-fragment layout) for the KV reduction.  A single 32 KB LDS staging
// buffer is reused for Kp then V so LDS does not cap occupancy.  B fragments
// are double-buffered across the K loop so loads for step f+1 issue before
// the four WMMAs of step f.
__global__ __launch_bounds__(64) void k_passA(
    const float* __restrict__ x,
    const __bf16* __restrict__ WqP, const __bf16* __restrict__ WkP,
    const __bf16* __restrict__ WvP,
    const float* __restrict__ bq, const float* __restrict__ bk,
    const float* __restrict__ bv,
    __bf16* __restrict__ Qp, __bf16* __restrict__ KpA, __bf16* __restrict__ Vp) {
  __shared__ __align__(16) __bf16 SB[32 * kHD];  // 32 KB staging (Kp then V)

  const int lane  = threadIdx.x & 31;
  const int wave  = threadIdx.x >> 5;
  const int chunk = blockIdx.x;
  const int n0    = chunk * 32 + wave * 16;
  const int col   = lane & 15;
  const int hi    = lane >> 4;

  // Preload 8 A-fragments of x (K = 256), converting f32 -> bf16.
  v16bf a[8];
  {
    int r = n0 + col;
    if (r >= kN) r = kN - 1;
    const float* xr = x + (size_t)r * kC;
#pragma unroll
    for (int f = 0; f < 8; ++f) {
      int kb = f * 32 + hi * 8;
      float4 p0 = *(const float4*)(xr + kb);
      float4 p1 = *(const float4*)(xr + kb + 4);
      float4 p2 = *(const float4*)(xr + kb + 16);
      float4 p3 = *(const float4*)(xr + kb + 20);
      v16bf av;
      av[0]  = (__bf16)p0.x; av[1]  = (__bf16)p0.y; av[2]  = (__bf16)p0.z; av[3]  = (__bf16)p0.w;
      av[4]  = (__bf16)p1.x; av[5]  = (__bf16)p1.y; av[6]  = (__bf16)p1.z; av[7]  = (__bf16)p1.w;
      av[8]  = (__bf16)p2.x; av[9]  = (__bf16)p2.y; av[10] = (__bf16)p2.z; av[11] = (__bf16)p2.w;
      av[12] = (__bf16)p3.x; av[13] = (__bf16)p3.y; av[14] = (__bf16)p3.z; av[15] = (__bf16)p3.w;
      a[f] = av;
    }
  }

  const unsigned short* SBS = (const unsigned short*)SB;
  const __bf16* Wsel[3] = {WqP, WkP, WvP};

#pragma unroll
  for (int T = 0; T < 3; ++T) {
    const __bf16* Wp = Wsel[T];
    for (int ct = 0; ct < 8; ++ct) {           // 64 output columns per pass
      v8f acc[4];
#pragma unroll
      for (int dt = 0; dt < 4; ++dt) acc[dt] = v8zero();

      v16bf bcur[4], bnxt[4];
#pragma unroll
      for (int dt = 0; dt < 4; ++dt)
        bcur[dt] = load_packed_frag(Wp, (long)0 * kNTW + ct * 4 + dt, lane);
#pragma unroll
      for (int f = 0; f < 8; ++f) {
        if (f < 7) {
#pragma unroll
          for (int dt = 0; dt < 4; ++dt)
            bnxt[dt] = load_packed_frag(Wp, (long)(f + 1) * kNTW + ct * 4 + dt, lane);
        }
#pragma unroll
        for (int dt = 0; dt < 4; ++dt)
          acc[dt] = WMMA_BF16(a[f], bcur[dt], acc[dt]);
#pragma unroll
        for (int dt = 0; dt < 4; ++dt) bcur[dt] = bnxt[dt];
      }

      // epilogue
#pragma unroll
      for (int dt = 0; dt < 4; ++dt) {
        int j = ct * 64 + dt * 16 + col;
#pragma unroll
        for (int v = 0; v < 8; ++v) {
          int M = v + hi * 8;
          int nrow = n0 + M;
          float val = acc[dt][v];
          if (T == 0) {
            val = __expf(val + bq[j]) + kEPS;
            if (nrow < kN) Qp[(size_t)nrow * kHD + j] = (__bf16)val;
          } else if (T == 1) {
            val = __expf(val + bk[j]) + kEPS;
            SB[(wave * 16 + M) * kHD + j] = (__bf16)val;
          } else {
            val = val + bv[j];
            SB[(wave * 16 + M) * kHD + j] = (__bf16)val;
          }
        }
      }
    }

    if (T == 1) {
      // Repack Kp as A-fragments (rows = feature m, K = node index).
      __syncthreads();
      for (int it = threadIdx.x; it < 32 * 32 * 8; it += 64) {
        int w = it & 7, ln = (it >> 3) & 31, tile = it >> 8;  // tile = h*4 + mt
        int c2 = ln & 15, h2 = ln >> 4;
        int off = (h2 == 0) ? ((w < 4) ? 2 * w : 2 * w + 8)
                            : ((w < 4) ? 2 * w + 8 : 2 * w + 16);
        int j = tile * 16 + c2;                                // h*64 + mt*16 + m
        int g = chunk * 32 + off;
        unsigned int lo = (g     < kN) ? SBS[off * kHD + j]       : 0u;
        unsigned int hv = (g + 1 < kN) ? SBS[(off + 1) * kHD + j] : 0u;
        ((unsigned int*)KpA)[((size_t)tile * kNCH + chunk) * 256 + ln * 8 + w] =
            lo | (hv << 16);
      }
      __syncthreads();  // SB will be overwritten by the V stage
    } else if (T == 2) {
      // Repack V as B-fragments (K = node index, cols = d).
      __syncthreads();
      for (int it = threadIdx.x; it < 32 * 32 * 8; it += 64) {
        int w = it & 7, ln = (it >> 3) & 31, tile = it >> 8;  // tile = h*4 + dt
        int c2 = ln & 15, h2 = ln >> 4;
        int off = h2 * 8 + 2 * w + ((w < 4) ? 0 : 8);
        int j = tile * 16 + c2;
        int g = chunk * 32 + off;
        unsigned int lo = (g     < kN) ? SBS[off * kHD + j]       : 0u;
        unsigned int hv = (g + 1 < kN) ? SBS[(off + 1) * kHD + j] : 0u;
        ((unsigned int*)Vp)[((size_t)tile * kNCH + chunk) * 256 + ln * 8 + w] =
            lo | (hv << 16);
      }
    }
  }
}

// ---------------------------------------------------------------------------
// Pass B: KV[h] = Kp_h^T @ V_h  and Kp_sum via an all-ones B fragment.
// Fragments for chunk c+1 are prefetched while chunk c's WMMAs execute.
__global__ __launch_bounds__(128) void k_passB(
    const __bf16* __restrict__ KpA, const __bf16* __restrict__ Vp,
    float* __restrict__ KVacc, float* __restrict__ Ksum, int cpb) {
  const int lane = threadIdx.x & 31;
  const int mt   = threadIdx.x >> 5;   // wave -> m tile
  const int h    = blockIdx.y;
  int c0 = blockIdx.x * cpb;
  int c1 = c0 + cpb; if (c1 > kNCH) c1 = kNCH;

  v16bf ones;
#pragma unroll
  for (int i = 0; i < 16; ++i) ones[i] = (__bf16)1.0f;

  v8f acc[4];
#pragma unroll
  for (int dt = 0; dt < 4; ++dt) acc[dt] = v8zero();
  v8f accS = v8zero();

  const long baseA = (long)(h * 4 + mt) * kNCH;
  const long baseB = (long)(h * 4) * kNCH;

  // software-pipelined stream over node chunks
  v16bf af, bf[4];
  if (c0 < c1) {
    af = load_packed_frag(KpA, baseA + c0, lane);
#pragma unroll
    for (int dt = 0; dt < 4; ++dt)
      bf[dt] = load_packed_frag(Vp, baseB + (long)dt * kNCH + c0, lane);
  }
  for (int c = c0; c < c1; ++c) {
    v16bf afn = af, bfn0 = bf[0], bfn1 = bf[1], bfn2 = bf[2], bfn3 = bf[3];
    if (c + 1 < c1) {
      afn  = load_packed_frag(KpA, baseA + c + 1, lane);
      bfn0 = load_packed_frag(Vp, baseB + 0 * (long)kNCH + c + 1, lane);
      bfn1 = load_packed_frag(Vp, baseB + 1 * (long)kNCH + c + 1, lane);
      bfn2 = load_packed_frag(Vp, baseB + 2 * (long)kNCH + c + 1, lane);
      bfn3 = load_packed_frag(Vp, baseB + 3 * (long)kNCH + c + 1, lane);
    }
    acc[0] = WMMA_BF16(af, bf[0], acc[0]);
    acc[1] = WMMA_BF16(af, bf[1], acc[1]);
    acc[2] = WMMA_BF16(af, bf[2], acc[2]);
    acc[3] = WMMA_BF16(af, bf[3], acc[3]);
    accS   = WMMA_BF16(af, ones, accS);
    af = afn; bf[0] = bfn0; bf[1] = bfn1; bf[2] = bfn2; bf[3] = bfn3;
  }

  const int colD = lane & 15, hi = lane >> 4;
#pragma unroll
  for (int dt = 0; dt < 4; ++dt)
#pragma unroll
    for (int v = 0; v < 8; ++v) {
      int m = mt * 16 + v + hi * 8;
      atomicAdd(&KVacc[((size_t)h * 64 + m) * 64 + dt * 16 + colD], acc[dt][v]);
    }
  if (colD == 0) {
#pragma unroll
    for (int v = 0; v < 8; ++v)
      atomicAdd(&Ksum[h * 64 + mt * 16 + v + hi * 8], accS[v]);
  }
}

// Build packed-B fragments of [KV | Kp_sum] in bf16 (5 d-tiles, 2 k-chunks).
__global__ void k_kvfin(const float* __restrict__ KVacc,
                        const float* __restrict__ Ksum,
                        __bf16* __restrict__ KVp) {
  int id = blockIdx.x * blockDim.x + threadIdx.x;
  if (id >= kH * 5 * 2 * 512) return;
  int t = id & 15, lane = (id >> 4) & 31, kc = (id >> 9) & 1;
  int rest = id >> 10;
  int dt = rest % 5, h = rest / 5;
  int m  = kc * 32 + (lane >> 4) * 8 + (t & 7) + (t >> 3) * 16;
  int dl = lane & 15;
  float val;
  if (dt < 4)      val = KVacc[((size_t)h * 64 + m) * 64 + dt * 16 + dl];
  else             val = (dl == 0) ? Ksum[h * 64 + m] : 0.f;
  KVp[id] = (__bf16)val;
}

// ---------------------------------------------------------------------------
// Pass C: Z = (Qp@KV)/(Qp.Kp_sum + eps) per head, then out = Z@Wo + b.
__global__ __launch_bounds__(128) void k_passC(
    const __bf16* __restrict__ Qp, const __bf16* __restrict__ KVp,
    const __bf16* __restrict__ WoP, const float* __restrict__ ob,
    float* __restrict__ out) {
  __shared__ __align__(16) __bf16 Z[4 * 16 * kHD];  // 64 KB, 16 KB per wave

  const int lane = threadIdx.x & 31;
  const int wave = threadIdx.x >> 5;
  const int n0   = blockIdx.x * 64 + wave * 16;
  const int col  = lane & 15;
  const int hi   = lane >> 4;

  int r = n0 + col;
  if (r >= kN) r = kN - 1;
  const __bf16* qrow = Qp + (size_t)r * kHD;
  __bf16* Zw = &Z[wave * 16 * kHD];

  for (int h = 0; h < kH; ++h) {
    v16bf a0 = load_afrag_row(qrow, h * 64, hi);
    v16bf a1 = load_afrag_row(qrow, h * 64 + 32, hi);
    v8f accZ[5];
#pragma unroll
    for (int dt = 0; dt < 5; ++dt) {
      v16bf b0 = load_packed_frag(KVp, (long)(h * 5 + dt) * 2 + 0, lane);
      v16bf b1 = load_packed_frag(KVp, (long)(h * 5 + dt) * 2 + 1, lane);
      v8f z = v8zero();
      z = WMMA_BF16(a0, b0, z);
      z = WMMA_BF16(a1, b1, z);
      accZ[dt] = z;
    }
    // Denominator lives in column 0 of tile 4 (lanes 0 and 16); broadcast.
    float den[8];
#pragma unroll
    for (int v = 0; v < 8; ++v)
      den[v] = __shfl(accZ[4][v], hi ? 16 : 0, 32) + kEPS;
#pragma unroll
    for (int dt = 0; dt < 4; ++dt)
#pragma unroll
      for (int v = 0; v < 8; ++v) {
        float zv = accZ[dt][v] / den[v];
        Zw[(v + hi * 8) * kHD + h * 64 + dt * 16 + col] = (__bf16)zv;
      }
  }

  // out[16 x 64] = Z[16 x 512] @ Wo + b, with double-buffered Wo fragments.
  v8f accO[4];
#pragma unroll
  for (int dt = 0; dt < 4; ++dt) accO[dt] = v8zero();
  const __bf16* zrow = Zw + col * kHD;
  v16bf wcur[4], wnxt[4];
#pragma unroll
  for (int dt = 0; dt < 4; ++dt)
    wcur[dt] = load_packed_frag(WoP, (long)dt, lane);
#pragma unroll
  for (int kt = 0; kt < 16; ++kt) {
    v16bf af = load_afrag_row(zrow, kt * 32, hi);
    if (kt < 15) {
#pragma unroll
      for (int dt = 0; dt < 4; ++dt)
        wnxt[dt] = load_packed_frag(WoP, (long)(kt + 1) * 4 + dt, lane);
    }
#pragma unroll
    for (int dt = 0; dt < 4; ++dt)
      accO[dt] = WMMA_BF16(af, wcur[dt], accO[dt]);
#pragma unroll
    for (int dt = 0; dt < 4; ++dt) wcur[dt] = wnxt[dt];
  }
#pragma unroll
  for (int dt = 0; dt < 4; ++dt) {
    float bias = ob[dt * 16 + col];
#pragma unroll
    for (int v = 0; v < 8; ++v) {
      int nn = n0 + v + hi * 8;
      if (nn < kN) out[(size_t)nn * 64 + dt * 16 + col] = accO[dt][v] + bias;
    }
  }
}

// ---------------------------------------------------------------------------
extern "C" void kernel_launch(void* const* d_in, const int* in_sizes, int n_in,
                              void* d_out, int out_size, void* d_ws, size_t ws_size,
                              hipStream_t stream) {
  const float* x    = (const float*)d_in[0];
  const float* Wq_w = (const float*)d_in[1];
  const float* Wq_b = (const float*)d_in[2];
  const float* Wk_w = (const float*)d_in[3];
  const float* Wk_b = (const float*)d_in[4];
  const float* Wv_w = (const float*)d_in[5];
  const float* Wv_b = (const float*)d_in[6];
  const float* Wo_w = (const float*)d_in[7];
  const float* Wo_b = (const float*)d_in[8];
  const float* proj = (const float*)d_in[9];
  float* out = (float*)d_out;

  char* ws = (char*)d_ws;
  size_t off = 0;
  auto alloc = [&](size_t bytes) -> void* {
    void* p = ws + off;
    off += (bytes + 255) & ~(size_t)255;
    return p;
  };

  __bf16* WqP   = (__bf16*)alloc((size_t)kC * kHD * 2);
  __bf16* WkP   = (__bf16*)alloc((size_t)kC * kHD * 2);
  __bf16* WvP   = (__bf16*)alloc((size_t)kC * kHD * 2);
  __bf16* WoP   = (__bf16*)alloc((size_t)kHD * kD * 2);
  float*  bq    = (float*)alloc(kHD * 4);
  float*  bk    = (float*)alloc(kHD * 4);
  __bf16* Qp    = (__bf16*)alloc((size_t)kN * kHD * 2);
  __bf16* KpA   = (__bf16*)alloc((size_t)32 * kNCH * 512 * 2);
  __bf16* Vp    = (__bf16*)alloc((size_t)32 * kNCH * 512 * 2);
  float*  KVacc = (float*)alloc((size_t)kH * 64 * 64 * 4);
  float*  Ksum  = (float*)alloc(kH * 64 * 4);
  __bf16* KVp   = (__bf16*)alloc((size_t)kH * 5 * 2 * 512 * 2);

  // Zero reduction accumulators (ws is not re-poisoned between replays).
  k_zero<<<(kH * 64 * 64 + 255) / 256, 256, 0, stream>>>(KVacc, kH * 64 * 64);
  k_zero<<<(kH * 64 + 255) / 256, 256, 0, stream>>>(Ksum, kH * 64);

  // Fold proj into Wq/Wk; pack all weights for WMMA B-fragment loads.
  int wthreads = kC * kHD;
  k_prep_wproj<<<(wthreads + 255) / 256, 256, 0, stream>>>(Wq_w, Wq_b, proj, WqP, bq);
  k_prep_wproj<<<(wthreads + 255) / 256, 256, 0, stream>>>(Wk_w, Wk_b, proj, WkP, bk);
  k_pack_w<<<(wthreads + 255) / 256, 256, 0, stream>>>(Wv_w, WvP, kC, kHD);
  k_pack_w<<<(kHD * kD + 255) / 256, 256, 0, stream>>>(Wo_w, WoP, kHD, kD);

  // Pass A: Qp / Kp / V.
  k_passA<<<kNCH, 64, 0, stream>>>(x, WqP, WkP, WvP, bq, bk, Wv_b, Qp, KpA, Vp);

  // Pass B: KV and Kp_sum reduction.
  const int cpb = 32;
  dim3 gridB((kNCH + cpb - 1) / cpb, kH);
  k_passB<<<gridB, 128, 0, stream>>>(KpA, Vp, KVacc, Ksum, cpb);

  // Finalize KV into packed bf16 fragments (with Kp_sum as 5th tile).
  k_kvfin<<<(kH * 5 * 2 * 512 + 255) / 256, 256, 0, stream>>>(KVacc, Ksum, KVp);

  // Pass C: numerator/denominator, normalize, output projection.
  k_passC<<<(kN + 63) / 64, 128, 0, stream>>>(Qp, KVp, WoP, Wo_b, out);
}
